// FlashSVDSwiGLUMLP_28140625723484
// MI455X (gfx1250) — matrix operand order repaired and checked by reading, not compiled, not run on hardware
//
#include <hip/hip_runtime.h>

// ---------------------------------------------------------------------------
// FlashSVD SwiGLU MLP for gfx1250 (MI455X): bf16 WMMA + async-LDS pipeline.
//   P = x*U1 ; H = P*V1+b1 ; act = silu(gate)*up ; Z = act*U2 ; out = Z*V2+b2
// GEMMs: V_WMMA_F32_16X16X32_BF16, f32 accumulate.
// Staging: GLOBAL_LOAD_ASYNC_TO_LDS_B128 (ASYNCcnt), double-buffered so the
// DMA for K-tile i+1 overlaps the WMMAs of K-tile i; one barrier per step.
// ---------------------------------------------------------------------------

typedef __bf16 bf16;
typedef __attribute__((ext_vector_type(16))) __bf16 v16bf;
typedef __attribute__((ext_vector_type(8)))  float  v8f;
typedef __attribute__((ext_vector_type(4)))  float  v4f;

#define HIDDEN_DIM 4096
#define INTER_DIM  11008
#define RANK_DIM   1024
#define TOKENS     4096   // B*S = 2*2048

union FragU { v16bf v; v4f h[2]; };

__device__ __forceinline__ v16bf ld_frag(const v4f* s, int i0, int i1) {
  FragU u;
  u.h[0] = s[i0];
  u.h[1] = s[i1];
  return u.v;
}

__device__ __forceinline__ v8f wmma_bf16(v16bf a, v16bf b, v8f c) {
  // (neg_a, A, neg_b, B, c_mod, C, reuse_a, reuse_b)
  return __builtin_amdgcn_wmma_f32_16x16x32_bf16(false, a, false, b,
                                                 (short)0, c, false, false);
}

// Raw LDS byte offset of a __shared__ object: ISA aperture rule says the low
// 32 bits of the flat address of an LDS location are the LDS offset.
__device__ __forceinline__ uint32_t lds_off(const void* p) {
  return (uint32_t)(size_t)p;
}

// CDNA5 async DMA: 16 bytes global -> LDS per lane, tracked by ASYNCcnt.
__device__ __forceinline__ void async_b128(uint32_t lds, const bf16* g) {
  uint64_t ga = (uint64_t)g;
  asm volatile("global_load_async_to_lds_b128 %0, %1, off"
               :: "v"(lds), "v"(ga) : "memory");
}

__device__ __forceinline__ void wait_async0() {
  asm volatile("s_wait_asynccnt 0x0" ::: "memory");
}

// ---------------------------------------------------------------------------
// Elementwise f32 -> bf16
// ---------------------------------------------------------------------------
__global__ void cvt_f32_to_bf16(const float* __restrict__ in,
                                bf16* __restrict__ out, size_t n) {
  size_t i = (size_t)blockIdx.x * blockDim.x + threadIdx.x;
  size_t stride = (size_t)gridDim.x * blockDim.x;
  for (; i < n; i += stride) out[i] = (bf16)in[i];
}

// ---------------------------------------------------------------------------
// Transpose + convert: W[K][N] (f32, row-major) -> WT[N][K] (bf16, row-major).
// WMMA's B-operand wants per-lane K-contiguous data, so weights go in as W^T.
// ---------------------------------------------------------------------------
__global__ void transpose_cvt_f32_to_bf16(const float* __restrict__ W,
                                          bf16* __restrict__ WT,
                                          int K, int N) {
  size_t total = (size_t)K * N;
  size_t i = (size_t)blockIdx.x * blockDim.x + threadIdx.x;
  size_t stride = (size_t)gridDim.x * blockDim.x;
  for (; i < total; i += stride) {
    int k = (int)(i / N);
    int n = (int)(i % N);
    WT[(size_t)n * K + k] = (bf16)W[i];
  }
}

// ---------------------------------------------------------------------------
// Tiled bf16 WMMA GEMM:  C[M][N] = A[M][K] * BT[N][K]^T  (+ bias)
// BM=128, BN=128, BK=32; 256 threads = 8 wave32s in a 2(M) x 4(N) grid.
// Each wave owns a 64x32 tile -> 4x2 accumulators, 8 WMMAs per K-step.
// LDS rows padded to 80B (5 x v4f); double-buffered async staging.
// ---------------------------------------------------------------------------
template<bool BIAS, bool OUT_BF16>
__global__ __launch_bounds__(256) void gemm_bf16_wmma(
    const bf16* __restrict__ A,     // [M][K]
    const bf16* __restrict__ BT,    // [N][K]
    const float* __restrict__ bias, // [N] or nullptr
    void* __restrict__ Cout,        // [M][N] (bf16 or f32)
    int M, int N, int K) {
  constexpr int BM = 128, BN = 128, BK = 32;
  (void)M;
  __shared__ v4f As4[2][BM * 5];
  __shared__ v4f Bs4[2][BN * 5];
  constexpr uint32_t ABUF = sizeof(As4[0]);   // 10240 B
  constexpr uint32_t BBUF = sizeof(Bs4[0]);   // 10240 B

  const int t     = threadIdx.x;
  const int lane  = t & 31;
  const int wave  = t >> 5;
  const int waveM = wave >> 2;   // 0..1 -> 64 rows each
  const int waveN = wave & 3;    // 0..3 -> 32 cols each
  const int lm    = lane & 15;   // fragment row/col within 16
  const int kh    = lane >> 4;   // K-half select
  const int m0    = blockIdx.y * BM;
  const int n0    = blockIdx.x * BN;

  // Cooperative staging: each thread DMAs 2 x 16B for A and for B per tile.
  const int srow = t >> 1;
  const int sc   = (t & 1) * 2;                 // chunk pair within 64B row
  const bf16* gA = A  + (size_t)(m0 + srow) * K + sc * 8;
  const bf16* gB = BT + (size_t)(n0 + srow) * K + sc * 8;
  const uint32_t aO = lds_off(&As4[0][srow * 5 + sc]);
  const uint32_t bO = lds_off(&Bs4[0][srow * 5 + sc]);

  // Prologue: async-fill buffer 0 with K-tile 0.
  async_b128(aO,      gA);
  async_b128(aO + 16, gA + 8);
  async_b128(bO,      gB);
  async_b128(bO + 16, gB + 8);

  v8f acc[4][2] = {};
  const int nk = K / BK;
  for (int i = 0; i < nk; ++i) {
    wait_async0();        // my DMA writes for tile i have landed in LDS
    __syncthreads();      // everyone's have; everyone done reading buf i^1

    if (i + 1 < nk) {     // kick off tile i+1 into the other buffer
      const uint32_t sw = (uint32_t)((i + 1) & 1);
      const bf16* ga = gA + (size_t)(i + 1) * BK;
      const bf16* gb = gB + (size_t)(i + 1) * BK;
      async_b128(aO + sw * ABUF,      ga);
      async_b128(aO + sw * ABUF + 16, ga + 8);
      async_b128(bO + sw * BBUF,      gb);
      async_b128(bO + sw * BBUF + 16, gb + 8);
      if (i + 2 < nk) {   // warm L2 one more tile ahead
        __builtin_prefetch(gA + (size_t)(i + 2) * BK, 0, 1);
        __builtin_prefetch(gB + (size_t)(i + 2) * BK, 0, 1);
      }
    }

    const v4f* As = As4[i & 1];
    const v4f* Bs = Bs4[i & 1];
    const int bb0 = (waveN * 32 + 0 * 16 + lm) * 5 + kh * 2;
    const int bb1 = (waveN * 32 + 1 * 16 + lm) * 5 + kh * 2;
    v16bf b0 = ld_frag(Bs, bb0, bb0 + 1);
    v16bf b1 = ld_frag(Bs, bb1, bb1 + 1);
#pragma unroll
    for (int mt = 0; mt < 4; ++mt) {
      const int ab = (waveM * 64 + mt * 16 + lm) * 5 + kh * 2;
      v16bf a = ld_frag(As, ab, ab + 1);
      acc[mt][0] = wmma_bf16(a, b0, acc[mt][0]);
      acc[mt][1] = wmma_bf16(a, b1, acc[mt][1]);
    }
  }

  // C layout: VGPR r, lanes 0-15 -> M=r, N=lane; lanes 16-31 -> M=8+r
#pragma unroll
  for (int mt = 0; mt < 4; ++mt) {
#pragma unroll
    for (int nt = 0; nt < 2; ++nt) {
      const int col  = n0 + waveN * 32 + nt * 16 + lm;
      const int row0 = m0 + waveM * 64 + mt * 16 + kh * 8;
      float bv = 0.0f;
      if constexpr (BIAS) bv = bias[col];
#pragma unroll
      for (int r = 0; r < 8; ++r) {
        float v = acc[mt][nt][r] + bv;
        if constexpr (OUT_BF16)
          ((bf16*)Cout)[(size_t)(row0 + r) * N + col] = (bf16)v;
        else
          ((float*)Cout)[(size_t)(row0 + r) * N + col] = v;
      }
    }
  }
}

// ---------------------------------------------------------------------------
// Fused SwiGLU GEMM: for output tile columns [n0, n0+128) computes both
//   gate = P * V1T[n]       and   up = P * V1T[n + INTER]
// then writes act = silu(gate+b1g) * (up+b1u) in bf16. H never hits HBM.
// BM=64, BN=128, BK=32; 8 waves in 2x4; each wave: 32x32 tile x 2 outputs.
// Same double-buffered async staging (3 tiles: A, B-gate, B-up).
// ---------------------------------------------------------------------------
__global__ __launch_bounds__(256) void gemm_swiglu_wmma(
    const bf16* __restrict__ P,    // [TOKENS][RANK]
    const bf16* __restrict__ V1T,  // [2*INTER][RANK]
    const float* __restrict__ b1,  // [2*INTER]
    bf16* __restrict__ act,        // [TOKENS][INTER]
    int K) {
  constexpr int BM = 64, BN = 128, BK = 32;
  __shared__ v4f As4[2][BM * 5];
  __shared__ v4f Bg4[2][BN * 5];
  __shared__ v4f Bu4[2][BN * 5];
  constexpr uint32_t ABUF = sizeof(As4[0]);   // 5120 B
  constexpr uint32_t GBUF = sizeof(Bg4[0]);   // 10240 B
  constexpr uint32_t UBUF = sizeof(Bu4[0]);   // 10240 B

  const int t     = threadIdx.x;
  const int lane  = t & 31;
  const int wave  = t >> 5;
  const int waveM = wave >> 2;   // 0..1 -> 32 rows each
  const int waveN = wave & 3;    // 0..3 -> 32 cols each
  const int lm    = lane & 15;
  const int kh    = lane >> 4;
  const int m0    = blockIdx.y * BM;
  const int n0    = blockIdx.x * BN;

  const int arow = t >> 2, ac = t & 3;        // A tile: 1 x 16B per thread
  const int brow = t >> 1, bc = (t & 1) * 2;  // each B tile: 2 x 16B per thread

  const bf16* gAp = P   + (size_t)(m0 + arow) * K + ac * 8;
  const bf16* gBg = V1T + (size_t)(n0 + brow) * K + bc * 8;
  const bf16* gBu = V1T + (size_t)(INTER_DIM + n0 + brow) * K + bc * 8;
  const uint32_t aO = lds_off(&As4[0][arow * 5 + ac]);
  const uint32_t gO = lds_off(&Bg4[0][brow * 5 + bc]);
  const uint32_t uO = lds_off(&Bu4[0][brow * 5 + bc]);

  // Prologue: tile 0 -> buffer 0.
  async_b128(aO,      gAp);
  async_b128(gO,      gBg);
  async_b128(gO + 16, gBg + 8);
  async_b128(uO,      gBu);
  async_b128(uO + 16, gBu + 8);

  v8f accg[2][2] = {};
  v8f accu[2][2] = {};
  const int nk = K / BK;
  for (int i = 0; i < nk; ++i) {
    wait_async0();
    __syncthreads();

    if (i + 1 < nk) {
      const uint32_t sw = (uint32_t)((i + 1) & 1);
      const bf16* ga = gAp + (size_t)(i + 1) * BK;
      const bf16* gg = gBg + (size_t)(i + 1) * BK;
      const bf16* gu = gBu + (size_t)(i + 1) * BK;
      async_b128(aO + sw * ABUF,      ga);
      async_b128(gO + sw * GBUF,      gg);
      async_b128(gO + sw * GBUF + 16, gg + 8);
      async_b128(uO + sw * UBUF,      gu);
      async_b128(uO + sw * UBUF + 16, gu + 8);
      if (i + 2 < nk) {
        __builtin_prefetch(gBg + (size_t)(i + 2) * BK, 0, 1);
        __builtin_prefetch(gBu + (size_t)(i + 2) * BK, 0, 1);
      }
    }

    const v4f* As = As4[i & 1];
    const v4f* Bg = Bg4[i & 1];
    const v4f* Bu = Bu4[i & 1];
    v16bf bg[2], bu[2];
#pragma unroll
    for (int nt = 0; nt < 2; ++nt) {
      const int bb = (waveN * 32 + nt * 16 + lm) * 5 + kh * 2;
      bg[nt] = ld_frag(Bg, bb, bb + 1);
      bu[nt] = ld_frag(Bu, bb, bb + 1);
    }
#pragma unroll
    for (int mt = 0; mt < 2; ++mt) {
      const int ab = (waveM * 32 + mt * 16 + lm) * 5 + kh * 2;
      v16bf a = ld_frag(As, ab, ab + 1);
#pragma unroll
      for (int nt = 0; nt < 2; ++nt) {
        accg[mt][nt] = wmma_bf16(a, bg[nt], accg[mt][nt]);
        accu[mt][nt] = wmma_bf16(a, bu[nt], accu[mt][nt]);
      }
    }
  }

#pragma unroll
  for (int mt = 0; mt < 2; ++mt) {
#pragma unroll
    for (int nt = 0; nt < 2; ++nt) {
      const int col  = n0 + waveN * 32 + nt * 16 + lm;
      const int row0 = m0 + waveM * 32 + mt * 16 + kh * 8;
      const float bvg = b1[col];
      const float bvu = b1[INTER_DIM + col];
#pragma unroll
      for (int r = 0; r < 8; ++r) {
        float g = accg[mt][nt][r] + bvg;
        float u = accu[mt][nt][r] + bvu;
        float s = g / (1.0f + __expf(-g));   // silu
        act[(size_t)(row0 + r) * INTER_DIM + col] = (bf16)(s * u);
      }
    }
  }
}

// ---------------------------------------------------------------------------
// Launch
// ---------------------------------------------------------------------------
extern "C" void kernel_launch(void* const* d_in, const int* in_sizes, int n_in,
                              void* d_out, int out_size, void* d_ws,
                              size_t ws_size, hipStream_t stream) {
  (void)in_sizes; (void)n_in; (void)out_size; (void)ws_size;

  const float* x  = (const float*)d_in[0];  // [T][H]
  const float* U1 = (const float*)d_in[1];  // [H][R]
  const float* V1 = (const float*)d_in[2];  // [R][2*I]
  const float* U2 = (const float*)d_in[3];  // [I][R]
  const float* V2 = (const float*)d_in[4];  // [R][H]
  const float* b1 = (const float*)d_in[5];  // [2*I]
  const float* b2 = (const float*)d_in[6];  // [H]
  float* out = (float*)d_out;               // [T][H]

  // bf16 workspace layout (element counts)
  bf16* ws = (bf16*)d_ws;
  size_t off = 0;
  bf16* xb  = ws + off; off += (size_t)TOKENS * HIDDEN_DIM;        // 16.8M
  bf16* U1T = ws + off; off += (size_t)RANK_DIM * HIDDEN_DIM;      // [R][H]
  bf16* V1T = ws + off; off += (size_t)2 * INTER_DIM * RANK_DIM;   // [2I][R]
  bf16* U2T = ws + off; off += (size_t)RANK_DIM * INTER_DIM;       // [R][I]
  bf16* V2T = ws + off; off += (size_t)HIDDEN_DIM * RANK_DIM;      // [H][R]
  bf16* Pb  = ws + off; off += (size_t)TOKENS * RANK_DIM;          // [T][R]
  bf16* Ab  = ws + off; off += (size_t)TOKENS * INTER_DIM;         // [T][I]
  bf16* Zb  = ws + off; off += (size_t)TOKENS * RANK_DIM;          // [T][R]

  // --- stage 0: convert / transpose-convert to bf16 ---
  cvt_f32_to_bf16<<<8192, 256, 0, stream>>>(x, xb,
      (size_t)TOKENS * HIDDEN_DIM);
  transpose_cvt_f32_to_bf16<<<4096, 256, 0, stream>>>(U1, U1T,
      HIDDEN_DIM, RANK_DIM);
  transpose_cvt_f32_to_bf16<<<4096, 256, 0, stream>>>(V1, V1T,
      RANK_DIM, 2 * INTER_DIM);
  transpose_cvt_f32_to_bf16<<<4096, 256, 0, stream>>>(U2, U2T,
      INTER_DIM, RANK_DIM);
  transpose_cvt_f32_to_bf16<<<4096, 256, 0, stream>>>(V2, V2T,
      RANK_DIM, HIDDEN_DIM);

  // --- stage 1: P = x * U1            (4096x1024, K=4096), bf16 out ---
  gemm_bf16_wmma<false, true><<<dim3(RANK_DIM / 128, TOKENS / 128),
                                256, 0, stream>>>(
      xb, U1T, nullptr, Pb, TOKENS, RANK_DIM, HIDDEN_DIM);

  // --- stage 2: act = silu(P*V1g+b1g) * (P*V1u+b1u)  (fused, K=1024) ---
  gemm_swiglu_wmma<<<dim3(INTER_DIM / 128, TOKENS / 64), 256, 0, stream>>>(
      Pb, V1T, b1, Ab, RANK_DIM);

  // --- stage 3: Z = act * U2          (4096x1024, K=11008), bf16 out ---
  gemm_bf16_wmma<false, true><<<dim3(RANK_DIM / 128, TOKENS / 128),
                                256, 0, stream>>>(
      Ab, U2T, nullptr, Zb, TOKENS, RANK_DIM, INTER_DIM);

  // --- stage 4: out = Z * V2 + b2     (4096x4096, K=1024), f32 out ---
  gemm_bf16_wmma<true, false><<<dim3(HIDDEN_DIM / 128, TOKENS / 128),
                                256, 0, stream>>>(
      Zb, V2T, b2, out, TOKENS, HIDDEN_DIM, RANK_DIM);
}
